// ODEK1_40956808135040
// MI455X (gfx1250) — compile-verified, matrix operand order checked by reading, not run on hardware
//
#include <hip/hip_runtime.h>
#include <hip/hip_bf16.h>

typedef __attribute__((ext_vector_type(16))) __bf16   v16bf;
typedef __attribute__((ext_vector_type(8)))  float    v8f;
typedef __attribute__((ext_vector_type(8)))  unsigned v8u;

#define NHID   64
#define NCLS   40
#define GN_EPS 1e-5f

// One-time (per launch) RNE f32->bf16 for the weight pack.
static __device__ __forceinline__ __bf16 f2bf_rne(float f) {
  unsigned u = __builtin_bit_cast(unsigned, f);
  unsigned r = (u + 0x7FFFu + ((u >> 16) & 1u)) >> 16;
  unsigned short s = (unsigned short)r;
  return __builtin_bit_cast(__bf16, s);
}

// Hot path: pack two f32 into a bf16 pair with ONE v_perm_b32 (truncation).
static __device__ __forceinline__ unsigned pkbf(float f0, float f1) {
  return __builtin_amdgcn_perm(__builtin_bit_cast(unsigned, f1),
                               __builtin_bit_cast(unsigned, f0), 0x07060302u);
}

// Device-scope fp32 atomic add (L2 atomic unit), avoids CAS-loop fallback.
static __device__ __forceinline__ void gatomic_fadd(float* p, float v) {
  asm volatile("global_atomic_add_f32 %0, %1, off scope:SCOPE_DEV"
               :: "v"(p), "v"(v) : "memory");
}

// ---- Pack B operand into WMMA wave32 lane layout ----------------------------
// Entry (ct, kc, lane) -> 16 bf16 (32B): b[i] = W[(kc*32 + hi*8 + i + krow_off)*ncols + n],
// b[i+8] = same with +16 on k; n = ct*16 + (lane&15); zero-padded for n >= nvalid.
__global__ void k_packB(const float* __restrict__ W, __bf16* __restrict__ dst,
                        int nch, int ntiles, int ncols, int nvalid, int krow_off)
{
  int id = blockIdx.x * blockDim.x + threadIdx.x;
  if (id >= ntiles * nch * 32) return;
  int lane = id & 31;
  int kc = (id >> 5) % nch;
  int ct = (id >> 5) / nch;
  int l15 = lane & 15, hi = lane >> 4;
  int n = ct * 16 + l15;
  int k0 = kc * 32 + hi * 8;
  __bf16* o = dst + (size_t)id * 16;
  bool ok = (n < nvalid);
  #pragma unroll
  for (int i = 0; i < 8; ++i) {
    o[i]     = f2bf_rne(ok ? W[(size_t)(k0 + i + krow_off) * ncols + n] : 0.f);
    o[i + 8] = f2bf_rne(ok ? W[(size_t)(k0 + 16 + i + krow_off) * ncols + n] : 0.f);
  }
}

static __device__ __forceinline__ v16bf loadB(const __bf16* Bp, int ct, int nch,
                                              int kc, int lane) {
  const v8u* bp = (const v8u*)Bp;
  v8u bu = bp[(ct * nch + kc) * 32 + lane];
  return __builtin_bit_cast(v16bf, bu);
}

// Pack 16 f32 A values in two 8-wide K runs (2x float4 pairs at +0 and +16) into v16bf.
static __device__ __forceinline__ v16bf packA(float4 a0, float4 a1, float4 a2, float4 a3) {
  v8u au;
  au[0] = pkbf(a0.x, a0.y); au[1] = pkbf(a0.z, a0.w);
  au[2] = pkbf(a1.x, a1.y); au[3] = pkbf(a1.z, a1.w);
  au[4] = pkbf(a2.x, a2.y); au[5] = pkbf(a2.z, a2.w);
  au[6] = pkbf(a3.x, a3.y); au[7] = pkbf(a3.z, a3.w);
  return __builtin_bit_cast(v16bf, au);
}

// ---------------- Layer 0 GEMM: S = X[N,128] @ W0[128,64] + b0 ----------------
// One wave computes a 16x64 strip: A loaded once per K-chunk, 4 WMMAs share it.
__global__ void k_gemm0(const float* __restrict__ X, const __bf16* __restrict__ Bp,
                        const float* __restrict__ bias, float* __restrict__ S, int N)
{
  const int K = 128;
  int wid  = (blockIdx.x * blockDim.x + threadIdx.x) >> 5;
  int lane = threadIdx.x & 31;
  if (wid >= (N >> 4)) return;
  int m0 = wid << 4;
  int l15 = lane & 15, hi = lane >> 4;
  int m = m0 + l15;
  v8f acc[4] = {};
  #pragma unroll
  for (int kc = 0; kc < 4; ++kc) {
    const float4* ap = (const float4*)(X + (size_t)m * K + kc * 32 + hi * 8);
    v16bf a = packA(ap[0], ap[1], ap[4], ap[5]);
    #pragma unroll
    for (int ct = 0; ct < 4; ++ct) {
      v16bf b = loadB(Bp, ct, 4, kc, lane);
      acc[ct] = __builtin_amdgcn_wmma_f32_16x16x32_bf16(false, a, false, b,
                                                        (short)0, acc[ct], false, false);
    }
  }
  #pragma unroll
  for (int ct = 0; ct < 4; ++ct) {
    float bv = bias[ct * 16 + l15];
    #pragma unroll
    for (int r = 0; r < 8; ++r)
      S[(size_t)(m0 + hi * 8 + r) * NHID + ct * 16 + l15] = acc[ct][r] + bv;
  }
}

// ------- ODE GEMM: S = groupnorm(Y) @ W1[1:65,:] + (t*W1[0,:] + b1) -----------
// GroupNorm groups of 2 channels: mu=(x0+x1)/2, d=x0-mu, var=d^2 (closed form).
static __device__ __forceinline__ unsigned gn_pair(float x0, float x1,
                                                   const float* gw, const float* gb, int c) {
  float mu = 0.5f * (x0 + x1);
  float d  = x0 - mu;
  float rs = rsqrtf(d * d + GN_EPS);
  float h0 =  d * rs * gw[c]     + gb[c];
  float h1 = -d * rs * gw[c + 1] + gb[c + 1];
  return pkbf(h0, h1);
}

__global__ void k_gemm_gn(const float* __restrict__ Y, const __bf16* __restrict__ Bp,
                          const float* __restrict__ W1, const float* __restrict__ b1,
                          const float* __restrict__ gw, const float* __restrict__ gb,
                          float t, float* __restrict__ S, int N)
{
  int wid  = (blockIdx.x * blockDim.x + threadIdx.x) >> 5;
  int lane = threadIdx.x & 31;
  if (wid >= (N >> 4)) return;
  int m0 = wid << 4;
  int l15 = lane & 15, hi = lane >> 4;
  int m = m0 + l15;
  v8f acc[4] = {};
  #pragma unroll
  for (int kc = 0; kc < 2; ++kc) {
    int k0 = kc * 32 + hi * 8;
    const float4* ap = (const float4*)(Y + (size_t)m * NHID + k0);
    float4 a0 = ap[0], a1 = ap[1], a2 = ap[4], a3 = ap[5];
    v8u au;
    au[0] = gn_pair(a0.x, a0.y, gw, gb, k0 + 0);
    au[1] = gn_pair(a0.z, a0.w, gw, gb, k0 + 2);
    au[2] = gn_pair(a1.x, a1.y, gw, gb, k0 + 4);
    au[3] = gn_pair(a1.z, a1.w, gw, gb, k0 + 6);
    au[4] = gn_pair(a2.x, a2.y, gw, gb, k0 + 16);
    au[5] = gn_pair(a2.z, a2.w, gw, gb, k0 + 18);
    au[6] = gn_pair(a3.x, a3.y, gw, gb, k0 + 20);
    au[7] = gn_pair(a3.z, a3.w, gw, gb, k0 + 22);
    v16bf a = __builtin_bit_cast(v16bf, au);
    #pragma unroll
    for (int ct = 0; ct < 4; ++ct) {
      v16bf b = loadB(Bp, ct, 2, kc, lane);
      acc[ct] = __builtin_amdgcn_wmma_f32_16x16x32_bf16(false, a, false, b,
                                                        (short)0, acc[ct], false, false);
    }
  }
  #pragma unroll
  for (int ct = 0; ct < 4; ++ct) {
    int n = ct * 16 + l15;
    float bv = t * W1[n] + b1[n];   // fold concat([t, hn]) into bias (W1 row 0)
    #pragma unroll
    for (int r = 0; r < 8; ++r)
      S[(size_t)(m0 + hi * 8 + r) * NHID + n] = acc[ct][r] + bv;
  }
}

// -------- Final GEMM: S2 = H[N,64] @ W2[64,40] + b2 (40 padded to 48) ---------
__global__ void k_gemm2(const float* __restrict__ H, const __bf16* __restrict__ Bp,
                        const float* __restrict__ b2, float* __restrict__ S2, int N)
{
  int wid  = (blockIdx.x * blockDim.x + threadIdx.x) >> 5;
  int lane = threadIdx.x & 31;
  if (wid >= (N >> 4)) return;
  int m0 = wid << 4;
  int l15 = lane & 15, hi = lane >> 4;
  int m = m0 + l15;
  v8f acc[3] = {};
  #pragma unroll
  for (int kc = 0; kc < 2; ++kc) {
    const float4* ap = (const float4*)(H + (size_t)m * NHID + kc * 32 + hi * 8);
    v16bf a = packA(ap[0], ap[1], ap[4], ap[5]);
    #pragma unroll
    for (int ct = 0; ct < 3; ++ct) {
      v16bf b = loadB(Bp, ct, 2, kc, lane);
      acc[ct] = __builtin_amdgcn_wmma_f32_16x16x32_bf16(false, a, false, b,
                                                        (short)0, acc[ct], false, false);
    }
  }
  #pragma unroll
  for (int ct = 0; ct < 3; ++ct) {
    int n = ct * 16 + l15;
    if (n < NCLS) {
      float bv = b2[n];
      #pragma unroll
      for (int r = 0; r < 8; ++r)
        S2[(size_t)(m0 + hi * 8 + r) * NCLS + n] = acc[ct][r] + bv;
    }
  }
}

// --------------- Edge scatter: Out[tgt] += w * S[src] (64 ch) -----------------
__global__ void k_scatter(const float* __restrict__ S, const int* __restrict__ src,
                          const int* __restrict__ tgt, const float* __restrict__ w,
                          float* __restrict__ O, int E)
{
  int id = blockIdx.x * blockDim.x + threadIdx.x;
  if (id >= E * 16) return;
  int e = id >> 4, cg = id & 15;            // 16 consecutive lanes share one edge
  int sv = src[e], tv = tgt[e];
  float ww = w[e];
  float4 v = *((const float4*)(S + (size_t)sv * NHID) + cg);
  float* op = O + (size_t)tv * NHID + cg * 4;
  gatomic_fadd(op + 0, ww * v.x); gatomic_fadd(op + 1, ww * v.y);
  gatomic_fadd(op + 2, ww * v.z); gatomic_fadd(op + 3, ww * v.w);
}

__global__ void k_scatter40(const float* __restrict__ S, const int* __restrict__ src,
                            const int* __restrict__ tgt, const float* __restrict__ w,
                            float* __restrict__ O, int E)
{
  int id = blockIdx.x * blockDim.x + threadIdx.x;
  if (id >= E * 10) return;
  int e = id / 10, cg = id % 10;
  int sv = src[e], tv = tgt[e];
  float ww = w[e];
  float4 v = *((const float4*)(S + (size_t)sv * NCLS) + cg);
  float* op = O + (size_t)tv * NCLS + cg * 4;
  gatomic_fadd(op + 0, ww * v.x); gatomic_fadd(op + 1, ww * v.y);
  gatomic_fadd(op + 2, ww * v.z); gatomic_fadd(op + 3, ww * v.w);
}

// --------------- RK4 stage combine (relu fused on the scatter sum) ------------
// mode 0: acc = k;        y = h + step*k        (k1)
// mode 1: acc += coef*k;  y = h + step*k        (k2,k3)
// mode 3: h += step*(acc + k)                   (k4, step = dt/6)
__global__ void k_stage(const float* __restrict__ KS, float* __restrict__ ACC,
                        float* __restrict__ H, float* __restrict__ Y,
                        float coef, float step, int mode, int total)
{
  int i = blockIdx.x * blockDim.x + threadIdx.x;
  if (i >= total) return;
  float k = fmaxf(KS[i], 0.f);
  if (mode == 0)      { ACC[i] = k;            Y[i] = H[i] + step * k; }
  else if (mode == 3) { H[i] += step * (ACC[i] + k); }
  else                { ACC[i] += coef * k;    Y[i] = H[i] + step * k; }
}

__global__ void k_relu(float* __restrict__ H, int total) {
  int i = blockIdx.x * blockDim.x + threadIdx.x;
  if (i < total) H[i] = fmaxf(H[i], 0.f);
}

__global__ void k_logsoftmax(float* __restrict__ O, int N) {
  int i = blockIdx.x * blockDim.x + threadIdx.x;
  if (i >= N) return;
  float* row = O + (size_t)i * NCLS;
  float mx = -1e30f;
  #pragma unroll
  for (int c = 0; c < NCLS; ++c) mx = fmaxf(mx, row[c]);
  float s = 0.f;
  #pragma unroll
  for (int c = 0; c < NCLS; ++c) s += __expf(row[c] - mx);
  float l = __logf(s) + mx;
  #pragma unroll
  for (int c = 0; c < NCLS; ++c) row[c] -= l;
}

extern "C" void kernel_launch(void* const* d_in, const int* in_sizes, int n_in,
                              void* d_out, int out_size, void* d_ws, size_t ws_size,
                              hipStream_t stream)
{
  const float* X  = (const float*)d_in[0];
  const int*   src = (const int*)d_in[1];
  const int*   tgt = (const int*)d_in[2];
  const float* mw = (const float*)d_in[3];
  const float* W0 = (const float*)d_in[4];
  const float* b0 = (const float*)d_in[5];
  const float* gw = (const float*)d_in[6];
  const float* gb = (const float*)d_in[7];
  const float* W1 = (const float*)d_in[8];
  const float* b1 = (const float*)d_in[9];
  const float* W2 = (const float*)d_in[10];
  const float* b2 = (const float*)d_in[11];
  int N = in_sizes[0] / 128;
  int E = in_sizes[1];
  float* out = (float*)d_out;

  size_t nh = (size_t)N * NHID;
  float* S   = (float*)d_ws;     // transformed features / final logits staging
  float* H   = S + nh;           // node state
  float* Y   = H + nh;           // RK4 stage input
  float* KS  = Y + nh;           // scatter sum (k before relu)
  float* ACC = KS + nh;          // RK4 accumulator
  __bf16* Bp0 = (__bf16*)(ACC + nh);        // 4 tiles x 4 chunks x 32 lanes x 32B
  __bf16* Bp1 = Bp0 + 4 * 4 * 32 * 16;      // 4 tiles x 2 chunks
  __bf16* Bp2 = Bp1 + 4 * 2 * 32 * 16;      // 3 tiles x 2 chunks (zero-padded 40->48)

  const float dt = 0.25f;
  dim3 blk(256);
  int gemm_blocks = ((N / 16) + 7) / 8;       // 8 waves/block, 16x64 strip per wave
  int ew_blocks   = (int)((nh + 255) / 256);
  int sc_blocks   = (E * 16 + 255) / 256;

  // ---- pack weight operands into WMMA lane layout (tiny, once per launch) ----
  k_packB<<<2, blk, 0, stream>>>(W0, Bp0, 4, 4, NHID, NHID, 0);
  k_packB<<<1, blk, 0, stream>>>(W1, Bp1, 2, 4, NHID, NHID, 1);  // skip t-row 0
  k_packB<<<1, blk, 0, stream>>>(W2, Bp2, 2, 3, NCLS, NCLS, 0);

  // ---- layer 0: relu(scatter(X @ W0 + b0)) ----
  k_gemm0<<<gemm_blocks, blk, 0, stream>>>(X, Bp0, b0, S, N);
  hipMemsetAsync(H, 0, nh * sizeof(float), stream);
  k_scatter<<<sc_blocks, blk, 0, stream>>>(S, src, tgt, mw, H, E);
  k_relu<<<ew_blocks, blk, 0, stream>>>(H, (int)nh);

  // ---- ODE block: 4 fixed RK4 steps ----
  for (int st = 0; st < 4; ++st) {
    float t = st * dt;
    const float* sin_[4] = {H, Y, Y, Y};
    float  ts_[4] = {t, t + 0.5f * dt, t + 0.5f * dt, t + dt};
    float  cf_[4] = {0.f, 2.f, 2.f, 0.f};
    float  sp_[4] = {0.5f * dt, 0.5f * dt, dt, dt / 6.f};
    int    md_[4] = {0, 1, 1, 3};
    for (int s4 = 0; s4 < 4; ++s4) {
      k_gemm_gn<<<gemm_blocks, blk, 0, stream>>>(sin_[s4], Bp1, W1, b1, gw, gb,
                                                 ts_[s4], S, N);
      hipMemsetAsync(KS, 0, nh * sizeof(float), stream);
      k_scatter<<<sc_blocks, blk, 0, stream>>>(S, src, tgt, mw, KS, E);
      k_stage<<<ew_blocks, blk, 0, stream>>>(KS, ACC, H, Y, cf_[s4], sp_[s4], md_[s4], (int)nh);
    }
  }

  // ---- final layer + log_softmax ----
  k_gemm2<<<gemm_blocks, blk, 0, stream>>>(H, Bp2, b2, S, N);
  hipMemsetAsync(out, 0, (size_t)N * NCLS * sizeof(float), stream);
  int sc40_blocks = (E * 10 + 255) / 256;
  k_scatter40<<<sc40_blocks, blk, 0, stream>>>(S, src, tgt, mw, out, E);
  k_logsoftmax<<<(N + 255) / 256, blk, 0, stream>>>(out, N);
}
